// GatedDeformConv_33767032881837
// MI455X (gfx1250) — matrix operand order repaired and checked by reading, not compiled, not run on hardware
//
#include <hip/hip_runtime.h>

// ---------------------------------------------------------------------------
// GatedDeformConv for MI455X (gfx1250): implicit-GEMM convs via
// v_wmma_f32_16x16x32_bf16, async global->LDS staging (double buffered),
// bilinear sampling fused into the deform GEMM.
// ---------------------------------------------------------------------------

typedef __bf16 bf16;
typedef __attribute__((ext_vector_type(16))) __bf16 v16bf;
typedef __attribute__((ext_vector_type(8)))  float  v8f;

#define BATCH 4
#define CH    128
#define HH    128
#define WWID  128
#define HWSZ  (HH * WWID)      // 16384 = 2^14
#define EPSV  1e-5f

union FragB16 { v16bf v; uint4 q[2]; };
union Pack8   { bf16 e[8]; uint4 q; };

__device__ inline v8f zero8() {
  v8f z;
#pragma unroll
  for (int i = 0; i < 8; ++i) z[i] = 0.0f;
  return z;
}

__device__ inline float sigmoidf_(float x) { return 1.0f / (1.0f + __expf(-x)); }

// Async copy 8B global -> LDS (ASYNCcnt tracked), and the matching wait.
__device__ inline void async_g2l_b64(unsigned ldsAddr, const void* gptr) {
  unsigned long long ga = (unsigned long long)gptr;
  asm volatile("global_load_async_to_lds_b64 %0, %1, off"
               :: "v"(ldsAddr), "v"(ga) : "memory");
}
__device__ inline void async_g2l_b128x2(unsigned ldsAddr, const void* gptr) {
  unsigned long long ga = (unsigned long long)gptr;
  asm volatile("global_load_async_to_lds_b128 %0, %1, off\n\t"
               "global_load_async_to_lds_b128 %0, %1, off offset:16"
               :: "v"(ldsAddr), "v"(ga) : "memory");
}
__device__ inline void wait_async0() {
  asm volatile("s_wait_asynccnt 0" ::: "memory");
}

// ---------------------------------------------------------------------------
// Packing kernels
// ---------------------------------------------------------------------------

// NCHW f32 -> NHWC bf16
__global__ void pack_x_kernel(const float* __restrict__ x, bf16* __restrict__ xt) {
  int i  = blockIdx.x * 256 + threadIdx.x;        // (b, c, hw), hw fastest
  int hw = i & (HWSZ - 1);
  int c  = (i >> 14) & (CH - 1);
  int b  = i >> 21;
  xt[((size_t)(b * HWSZ) + hw) * CH + c] = (bf16)x[i];
}

// w_attn + w_feat (OIHW f32) -> fragment-linear bf16 [t][kc][nt 0..15][lane][16]
__global__ void pack_wAF_kernel(const float* __restrict__ wa,
                                const float* __restrict__ wf,
                                bf16* __restrict__ out) {
  int i    = blockIdx.x * 256 + threadIdx.x;       // 294912
  int j    = i & 15;
  int lane = (i >> 4) & 31;
  int nt   = (i >> 9) & 15;
  int kc   = (i >> 13) & 3;
  int t    = i >> 15;
  int k    = ((lane >> 4) << 3) + (j & 7) + ((j >> 3) << 4);
  int n    = lane & 15;
  int ci   = kc * 32 + k;
  int co   = nt * 16 + n;
  float v = (co < 128) ? wa[(co * CH + ci) * 9 + t]
                       : wf[((co - 128) * CH + ci) * 9 + t];
  out[i] = (bf16)v;
}

// w_off (18 ch) + w_mask (9 ch) -> N=32 padded fragment-linear bf16
__global__ void pack_wOM_kernel(const float* __restrict__ wo,
                                const float* __restrict__ wm,
                                bf16* __restrict__ out) {
  int i    = blockIdx.x * 256 + threadIdx.x;       // 36864
  int j    = i & 15;
  int lane = (i >> 4) & 31;
  int nt   = (i >> 9) & 1;
  int kc   = (i >> 10) & 3;
  int t    = i >> 12;
  int k    = ((lane >> 4) << 3) + (j & 7) + ((j >> 3) << 4);
  int n    = lane & 15;
  int ci   = kc * 32 + k;
  int co   = nt * 16 + n;
  float v = 0.0f;
  if (co < 18)      v = wo[(co * CH + ci) * 9 + t];
  else if (co < 27) v = wm[((co - 18) * CH + ci) * 9 + t];
  out[i] = (bf16)v;
}

// w_org -> fragment-linear bf16, K ordering = tap-major (t*128 + ci)
__global__ void pack_wORG_kernel(const float* __restrict__ w,
                                 bf16* __restrict__ out) {
  int i    = blockIdx.x * 256 + threadIdx.x;       // 147456
  int j    = i & 15;
  int lane = (i >> 4) & 31;
  int nt   = (i >> 9) & 7;
  int kc   = (i >> 12) & 3;
  int t    = i >> 14;
  int k    = ((lane >> 4) << 3) + (j & 7) + ((j >> 3) << 4);
  int n    = lane & 15;
  int ci   = kc * 32 + k;
  int co   = nt * 16 + n;
  out[i] = (bf16)w[(co * CH + ci) * 9 + t];
}

// ---------------------------------------------------------------------------
// Fused attn + feat conv: implicit GEMM, M-tile=32, N=256 (attn|feat).
// Double-buffered async A staging; 36 K-chunks of 32.
// ---------------------------------------------------------------------------
__global__ __launch_bounds__(256) void conv_attn_feat_kernel(
    const bf16* __restrict__ xt, const bf16* __restrict__ wAF,
    const float* __restrict__ b_attn, const float* __restrict__ b_feat,
    float* __restrict__ attn, float* __restrict__ feat) {
  __shared__ bf16 sA[2][32 * 32];
  const int tid  = threadIdx.x;
  const int wave = tid >> 5, lane = tid & 31;
  const int posBase = blockIdx.x * 32;             // flat b*HW + hw
  const int b    = posBase >> 14;
  const int msub = wave & 1;
  const int ntBase = (wave >> 1) * 4;

  v8f acc[4];
#pragma unroll
  for (int i = 0; i < 4; ++i) acc[i] = zero8();

  const int sPos = tid >> 3;                       // 0..31 staged position
  const int sCh  = (tid & 7) * 4;                  // 4 bf16 per thread
  const int hw0  = (posBase & (HWSZ - 1)) + sPos;
  const int py = hw0 >> 7, px = hw0 & 127;
  const unsigned ldsBase = (unsigned)(size_t)(&sA[0][0]);
  const unsigned myLds   = (unsigned)((sPos * 32 + sCh) * 2);

  auto stage = [&](int cc, int buf) {
    const int t = cc >> 2, kc = cc & 3;
    const int sy = py + t / 3 - 1, sx = px + t % 3 - 1;
    const bool valid = (sy >= 0) & (sy < HH) & (sx >= 0) & (sx < WWID);
    if (valid) {
      const bf16* g = xt + ((size_t)(b * HWSZ) + (size_t)sy * WWID + sx) * CH
                         + kc * 32 + sCh;
      async_g2l_b64(ldsBase + (unsigned)buf * 2048u + myLds, g);
    } else {
      uint2 z; z.x = 0u; z.y = 0u;
      *(uint2*)(&sA[buf][sPos * 32 + sCh]) = z;
    }
  };

  stage(0, 0);
  wait_async0();
  __syncthreads();

  for (int cc = 0; cc < 36; ++cc) {
    const int buf = cc & 1;
    if (cc + 1 < 36) stage(cc + 1, buf ^ 1);       // overlap copy with compute

    FragB16 af;
    const int am = msub * 16 + (lane & 15);
    const int kg = (lane >> 4) * 8;                // A layout: K kg..kg+7 / 16+kg..
    af.q[0] = *(const uint4*)(&sA[buf][am * 32 + kg]);
    af.q[1] = *(const uint4*)(&sA[buf][am * 32 + 16 + kg]);

    const bf16* wB = wAF + (size_t)(cc * 16) * 512 + (size_t)lane * 16;
#pragma unroll
    for (int n = 0; n < 4; ++n) {
      FragB16 bfv;
      const bf16* p = wB + (size_t)(ntBase + n) * 512;
      bfv.q[0] = *(const uint4*)p;
      bfv.q[1] = *(const uint4*)(p + 8);
      acc[n] = __builtin_amdgcn_wmma_f32_16x16x32_bf16(
          false, af.v, false, bfv.v, (short)0, acc[n], false, false);
    }
    wait_async0();
    __syncthreads();
  }

  const int n16 = lane & 15, mhi = (lane >> 4) * 8;
#pragma unroll
  for (int n = 0; n < 4; ++n) {
    int coG = (ntBase + n) * 16 + n16;
    bool isAttn = coG < 128;
    int co = isAttn ? coG : coG - 128;
    float bias = isAttn ? b_attn[co] : b_feat[co];
#pragma unroll
    for (int v = 0; v < 8; ++v) {                  // C layout: M = mhi + v
      size_t pos = (size_t)(posBase + msub * 16 + mhi + v);
      float val = acc[n][v] + bias;
      if (isAttn) attn[pos * CH + co] = sigmoidf_(val);
      else        feat[pos * CH + co] = val;
    }
  }
}

// ---------------------------------------------------------------------------
// Offset + mask conv: N=32 (18 off, 9 mask, 5 pad), M-tile=128, async staging.
// ---------------------------------------------------------------------------
__global__ __launch_bounds__(256) void conv_off_mask_kernel(
    const bf16* __restrict__ xa, const bf16* __restrict__ wOM,
    const float* __restrict__ b_off, const float* __restrict__ b_mask,
    float* __restrict__ om) {
  __shared__ bf16 sA[2][128 * 32];
  const int tid = threadIdx.x;
  const int wave = tid >> 5, lane = tid & 31;
  const int posBase = blockIdx.x * 128;
  const int b = posBase >> 14;
  v8f acc[2];
  acc[0] = zero8(); acc[1] = zero8();

  const int sPos = tid >> 1;                       // 0..127
  const int sCh  = (tid & 1) * 16;                 // 16 bf16 per thread
  const int hw0 = (posBase & (HWSZ - 1)) + sPos;
  const int py = hw0 >> 7, px = hw0 & 127;
  const unsigned ldsBase = (unsigned)(size_t)(&sA[0][0]);
  const unsigned myLds   = (unsigned)((sPos * 32 + sCh) * 2);

  auto stage = [&](int cc, int buf) {
    const int t = cc >> 2, kc = cc & 3;
    const int sy = py + t / 3 - 1, sx = px + t % 3 - 1;
    const bool valid = (sy >= 0) & (sy < HH) & (sx >= 0) & (sx < WWID);
    if (valid) {
      const bf16* g = xa + ((size_t)(b * HWSZ) + (size_t)sy * WWID + sx) * CH
                         + kc * 32 + sCh;
      async_g2l_b128x2(ldsBase + (unsigned)buf * 8192u + myLds, g);
    } else {
      uint4 z; z.x = z.y = z.z = z.w = 0u;
      *(uint4*)(&sA[buf][sPos * 32 + sCh])     = z;
      *(uint4*)(&sA[buf][sPos * 32 + sCh + 8]) = z;
    }
  };

  stage(0, 0);
  wait_async0();
  __syncthreads();

  for (int cc = 0; cc < 36; ++cc) {
    const int buf = cc & 1;
    if (cc + 1 < 36) stage(cc + 1, buf ^ 1);

    FragB16 af;
    const int am = wave * 16 + (lane & 15);
    const int kg = (lane >> 4) * 8;
    af.q[0] = *(const uint4*)(&sA[buf][am * 32 + kg]);
    af.q[1] = *(const uint4*)(&sA[buf][am * 32 + 16 + kg]);

    const bf16* wB = wOM + (size_t)(cc * 2) * 512 + (size_t)lane * 16;
#pragma unroll
    for (int n = 0; n < 2; ++n) {
      FragB16 bfv;
      const bf16* p = wB + (size_t)n * 512;
      bfv.q[0] = *(const uint4*)p;
      bfv.q[1] = *(const uint4*)(p + 8);
      acc[n] = __builtin_amdgcn_wmma_f32_16x16x32_bf16(
          false, af.v, false, bfv.v, (short)0, acc[n], false, false);
    }
    wait_async0();
    __syncthreads();
  }

  const int n16 = lane & 15, mhi = (lane >> 4) * 8;
#pragma unroll
  for (int n = 0; n < 2; ++n) {
    int co = n * 16 + n16;
#pragma unroll
    for (int v = 0; v < 8; ++v) {
      size_t pos = (size_t)(posBase + wave * 16 + mhi + v);
      float val = acc[n][v];
      if (co < 18)      om[pos * 32 + co] = val + b_off[co];
      else if (co < 27) om[pos * 32 + co] = sigmoidf_(val + b_mask[co - 18]);
    }
  }
}

// ---------------------------------------------------------------------------
// Deform GEMM: bilinear sampling fused into the K loop (tap-major K=1152).
// M-tile=64, N=128; double-buffered LDS A so one barrier per chunk.
// ---------------------------------------------------------------------------
__global__ __launch_bounds__(256) void deform_gemm_kernel(
    const bf16* __restrict__ xa, const float* __restrict__ om,
    const bf16* __restrict__ wORG, const float* __restrict__ b_org,
    float* __restrict__ dconv) {
  __shared__ bf16 sA[2][64 * 32];
  __shared__ int   sY0[64], sX0[64];
  __shared__ float sWy[64], sWx[64], sM[64];

  const int tid = threadIdx.x;
  const int wave = tid >> 5, lane = tid & 31;
  const int posBase = blockIdx.x * 64;
  const int b = posBase >> 14;
  const int msub = wave & 3;
  const int ntBase = (wave >> 2) * 4;

  v8f acc[4];
#pragma unroll
  for (int i = 0; i < 4; ++i) acc[i] = zero8();

  const int gPos = tid >> 2;                       // 0..63 sampled position
  const int gCh  = (tid & 3) * 8;                  // 8 channels per thread

  for (int t = 0; t < 9; ++t) {
    if (tid < 64) {
      int pos = posBase + tid;
      int hw = pos & (HWSZ - 1);
      float dy = om[(size_t)pos * 32 + t * 2];
      float dx = om[(size_t)pos * 32 + t * 2 + 1];
      float yy = dy + (float)(hw >> 7) + (float)(t / 3) - 1.0f;
      float xx = dx + (float)(hw & 127) + (float)(t % 3) - 1.0f;
      float y0 = floorf(yy), x0 = floorf(xx);
      sY0[tid] = (int)y0; sX0[tid] = (int)x0;
      sWy[tid] = yy - y0; sWx[tid] = xx - x0;
      sM[tid]  = om[(size_t)pos * 32 + 18 + t];
    }
    __syncthreads();
    const int   y0 = sY0[gPos], x0 = sX0[gPos];
    const float wy = sWy[gPos], wx = sWx[gPos], mk = sM[gPos];

    for (int kc = 0; kc < 4; ++kc) {
      const int buf = kc & 1;
      float accf[8];
#pragma unroll
      for (int j = 0; j < 8; ++j) accf[j] = 0.0f;
#pragma unroll
      for (int corner = 0; corner < 4; ++corner) {
        int yi = y0 + (corner >> 1), xi = x0 + (corner & 1);
        float w = ((corner >> 1) ? wy : 1.0f - wy) *
                  ((corner & 1)  ? wx : 1.0f - wx);
        if (yi >= 0 && yi < HH && xi >= 0 && xi < WWID) {
          const bf16* p = xa + ((size_t)(b * HWSZ) + (size_t)yi * WWID + xi) * CH
                             + kc * 32 + gCh;
          uint4 q = *(const uint4*)p;
          const bf16* e = (const bf16*)&q;
#pragma unroll
          for (int j = 0; j < 8; ++j) accf[j] += w * (float)e[j];
        }
      }
      Pack8 o;
#pragma unroll
      for (int j = 0; j < 8; ++j) o.e[j] = (bf16)(accf[j] * mk);
      *(uint4*)(&sA[buf][gPos * 32 + gCh]) = o.q;
      __syncthreads();

      FragB16 af;
      const int am = msub * 16 + (lane & 15);
      const int kg = (lane >> 4) * 8;
      af.q[0] = *(const uint4*)(&sA[buf][am * 32 + kg]);
      af.q[1] = *(const uint4*)(&sA[buf][am * 32 + 16 + kg]);

      const bf16* wB = wORG + (size_t)((t * 4 + kc) * 8) * 512 + (size_t)lane * 16;
#pragma unroll
      for (int n = 0; n < 4; ++n) {
        FragB16 bfv;
        const bf16* p = wB + (size_t)(ntBase + n) * 512;
        bfv.q[0] = *(const uint4*)p;
        bfv.q[1] = *(const uint4*)(p + 8);
        acc[n] = __builtin_amdgcn_wmma_f32_16x16x32_bf16(
            false, af.v, false, bfv.v, (short)0, acc[n], false, false);
      }
    }
  }
  const int n16 = lane & 15, mhi = (lane >> 4) * 8;
#pragma unroll
  for (int n = 0; n < 4; ++n) {
    int co = (ntBase + n) * 16 + n16;
    float bias = b_org[co];
#pragma unroll
    for (int v = 0; v < 8; ++v) {
      size_t pos = (size_t)(posBase + msub * 16 + mhi + v);
      dconv[pos * CH + co] = acc[n][v] + bias;
    }
  }
}

// ---------------------------------------------------------------------------
// Instance-norm stats over HW per (b,c): stats[2*(b*128+c)] = {mu, rsqrt(var+eps)}
// ---------------------------------------------------------------------------
__global__ __launch_bounds__(256) void inorm_stats_kernel(
    const float* __restrict__ buf, float* __restrict__ stats) {
  __shared__ float sSum[256], sSq[256];
  int c = blockIdx.x & (CH - 1);
  int b = blockIdx.x >> 7;
  const float* p = buf + (size_t)(b * HWSZ) * CH + c;
  float s = 0.0f, q = 0.0f;
  for (int i = threadIdx.x; i < HWSZ; i += 256) {
    float v = p[(size_t)i * CH];
    s += v; q += v * v;
  }
  sSum[threadIdx.x] = s; sSq[threadIdx.x] = q;
  __syncthreads();
  for (int off = 128; off > 0; off >>= 1) {
    if (threadIdx.x < (unsigned)off) {
      sSum[threadIdx.x] += sSum[threadIdx.x + off];
      sSq[threadIdx.x]  += sSq[threadIdx.x + off];
    }
    __syncthreads();
  }
  if (threadIdx.x == 0) {
    float mu  = sSum[0] * (1.0f / (float)HWSZ);
    float var = sSq[0] * (1.0f / (float)HWSZ) - mu * mu;
    stats[2 * blockIdx.x]     = mu;
    stats[2 * blockIdx.x + 1] = rsqrtf(var + EPSV);
  }
}

// x_attned = lrelu(IN(feat)) * attn   -> bf16 NHWC
__global__ void fuse1_kernel(const float* __restrict__ feat,
                             const float* __restrict__ attn,
                             const float* __restrict__ statsF,
                             bf16* __restrict__ xa) {
  int i = blockIdx.x * 256 + threadIdx.x;          // B*HW*CH
  int c = i & (CH - 1);
  int b = i >> 21;
  int sc = b * CH + c;
  float v = (feat[i] - statsF[2 * sc]) * statsF[2 * sc + 1];
  v = v > 0.0f ? v : 0.2f * v;
  xa[i] = (bf16)(v * attn[i]);
}

// out(NCHW f32) = x_attned + lrelu(IN(dconv)) * (1 - attn)
__global__ void fuse2_kernel(const float* __restrict__ feat,
                             const float* __restrict__ attn,
                             const float* __restrict__ statsF,
                             const float* __restrict__ dconv,
                             const float* __restrict__ statsD,
                             float* __restrict__ out) {
  int j  = blockIdx.x * 256 + threadIdx.x;         // NCHW, hw fastest
  int hw = j & (HWSZ - 1);
  int c  = (j >> 14) & (CH - 1);
  int b  = j >> 21;
  size_t i = ((size_t)(b * HWSZ) + hw) * CH + c;
  int sc = b * CH + c;
  float a  = attn[i];
  float v1 = (feat[i] - statsF[2 * sc]) * statsF[2 * sc + 1];
  v1 = v1 > 0.0f ? v1 : 0.2f * v1;
  float xat = v1 * a;
  float v2 = (dconv[i] - statsD[2 * sc]) * statsD[2 * sc + 1];
  v2 = v2 > 0.0f ? v2 : 0.2f * v2;
  out[j] = xat + v2 * (1.0f - a);
}

// ---------------------------------------------------------------------------
extern "C" void kernel_launch(void* const* d_in, const int* in_sizes, int n_in,
                              void* d_out, int out_size, void* d_ws, size_t ws_size,
                              hipStream_t stream) {
  (void)in_sizes; (void)n_in; (void)out_size; (void)ws_size;
  const float* x      = (const float*)d_in[0];
  const float* w_attn = (const float*)d_in[1];
  const float* b_attn = (const float*)d_in[2];
  const float* w_feat = (const float*)d_in[3];
  const float* b_feat = (const float*)d_in[4];
  const float* w_org  = (const float*)d_in[5];
  const float* b_org  = (const float*)d_in[6];
  const float* w_off  = (const float*)d_in[7];
  const float* b_off  = (const float*)d_in[8];
  const float* w_mask = (const float*)d_in[9];
  const float* b_mask = (const float*)d_in[10];

  char* ws = (char*)d_ws;
  size_t off = 0;
  auto take = [&](size_t bytes) -> char* {
    char* p = ws + off;
    off += (bytes + 255) & ~(size_t)255;
    return p;
  };
  const size_t NE = (size_t)BATCH * HWSZ * CH;     // 8,388,608
  bf16*  xt    = (bf16*)take(NE * 2);
  bf16*  xa    = (bf16*)take(NE * 2);
  float* attn  = (float*)take(NE * 4);
  float* feat  = (float*)take(NE * 4);
  float* dconv = (float*)take(NE * 4);
  float* om    = (float*)take((size_t)BATCH * HWSZ * 32 * 4);
  bf16*  wAF   = (bf16*)take(294912 * 2);
  bf16*  wOM   = (bf16*)take(36864 * 2);
  bf16*  wORG  = (bf16*)take(147456 * 2);
  float* statsF = (float*)take((size_t)BATCH * CH * 2 * 4);
  float* statsD = (float*)take((size_t)BATCH * CH * 2 * 4);

  pack_x_kernel   <<<32768, 256, 0, stream>>>(x, xt);
  pack_wAF_kernel <<<1152,  256, 0, stream>>>(w_attn, w_feat, wAF);
  pack_wOM_kernel <<<144,   256, 0, stream>>>(w_off, w_mask, wOM);
  pack_wORG_kernel<<<576,   256, 0, stream>>>(w_org, wORG);

  conv_attn_feat_kernel<<<2048, 256, 0, stream>>>(xt, wAF, b_attn, b_feat, attn, feat);
  inorm_stats_kernel   <<<BATCH * CH, 256, 0, stream>>>(feat, statsF);
  fuse1_kernel         <<<32768, 256, 0, stream>>>(feat, attn, statsF, xa);

  conv_off_mask_kernel <<<512,  256, 0, stream>>>(xa, wOM, b_off, b_mask, om);
  deform_gemm_kernel   <<<1024, 256, 0, stream>>>(xa, om, wORG, b_org, dconv);
  inorm_stats_kernel   <<<BATCH * CH, 256, 0, stream>>>(dconv, statsD);

  fuse2_kernel<<<32768, 256, 0, stream>>>(feat, attn, statsF, dconv, statsD,
                                          (float*)d_out);
}
